// Attention_48146583388331
// MI455X (gfx1250) — matrix operand order repaired and checked by reading, not compiled, not run on hardware
//
#include <hip/hip_runtime.h>
#include <hip/hip_bf16.h>

typedef _Float16 half_t;
typedef half_t hv4  __attribute__((ext_vector_type(4)));
typedef half_t hv8  __attribute__((ext_vector_type(8)));
typedef half_t hv16 __attribute__((ext_vector_type(16)));
typedef float  fv8  __attribute__((ext_vector_type(8)));
typedef unsigned int uv4 __attribute__((ext_vector_type(4)));
typedef int iv4 __attribute__((ext_vector_type(4)));
typedef int iv8 __attribute__((ext_vector_type(8)));

#define BATCH 512
#define NQ    196      // window tokens
#define NP    224      // keys padded to 14*16
#define DMODEL 512
#define HEADS  8
#define HDIM   64
#define MROWS (BATCH * NQ)   // 100352

__device__ __forceinline__ hv16 cat8(hv8 lo, hv8 hi) {
    return __builtin_shufflevector(lo, hi, 0,1,2,3,4,5,6,7,8,9,10,11,12,13,14,15);
}

__device__ __forceinline__ fv8 wmma16(hv16 a, hv16 b, fv8 c) {
    // v_wmma_f32_16x16x32_f16: D = A(16x32) * B(32x16) + C
    return __builtin_amdgcn_wmma_f32_16x16x32_f16(false, a, false, b, (short)0, c, false, false);
}

// ---------------------------------------------------------------------------
// Kernel 1: QKV = x @ w_qkv + b_qkv  (f32 in, f16 out, head-major layout)
// ---------------------------------------------------------------------------
__global__ __launch_bounds__(256) void qkv_kernel(
    const float* __restrict__ x, const float* __restrict__ w,
    const float* __restrict__ bqkv,
    half_t* __restrict__ q_ws, half_t* __restrict__ k_ws, half_t* __restrict__ v_ws)
{
    __shared__ alignas(16) half_t As[128][40];   // 128 rows x 32 k (pad 40)
    __shared__ alignas(16) half_t Bt[128][40];   // 128 cols x 32 k (transposed)

    const int tid  = threadIdx.x;
    const int lane = tid & 31;
    const int wave = tid >> 5;
    const int wr = wave >> 1;        // 0..3 -> 32-row slab
    const int wc = wave & 1;         // 0..1 -> 64-col slab
    const int m0 = blockIdx.x * 128;
    const int n0 = blockIdx.y * 128;

    fv8 zero8 = {};
    fv8 acc[2][4];
#pragma unroll
    for (int rt = 0; rt < 2; ++rt)
#pragma unroll
        for (int ct = 0; ct < 4; ++ct) acc[rt][ct] = zero8;

    const int kb    = (lane & 16) ? 8 : 0;   // A-frag half-group offset
    const int koffb = (lane & 16);           // unused; B-frag uses kb, same as A
    const int lc    = lane & 15;

    for (int kk = 0; kk < DMODEL; kk += 32) {
        // prefetch next k-step tiles into GL2/WGP$ (global_prefetch_b8)
        if (kk + 32 < DMODEL) {
            int arow = (tid * 4) >> 3;
            __builtin_prefetch(x + (size_t)(m0 + arow) * DMODEL + kk + 32, 0, 1);
            int brow = (tid * 4) >> 5;
            __builtin_prefetch(w + (size_t)(kk + 32 + brow) * 1536 + n0, 0, 1);
        }
        // stage A: 128x32 f32 -> f16 (1024 float4 slots / 256 threads)
#pragma unroll
        for (int i = 0; i < 4; ++i) {
            int slot = tid + i * 256;
            int row = slot >> 3, cv = slot & 7;
            const float4 f = *(const float4*)(x + (size_t)(m0 + row) * DMODEL + kk + cv * 4);
            hv4 h4 = { (half_t)f.x, (half_t)f.y, (half_t)f.z, (half_t)f.w };
            *(hv4*)&As[row][cv * 4] = h4;
        }
        // stage B transposed: rows k, cols n -> Bt[n][k]
#pragma unroll
        for (int i = 0; i < 4; ++i) {
            int slot = tid + i * 256;
            int kr = slot >> 5, nv = slot & 31;
            const float4 f = *(const float4*)(w + (size_t)(kk + kr) * 1536 + n0 + nv * 4);
            Bt[nv * 4 + 0][kr] = (half_t)f.x;
            Bt[nv * 4 + 1][kr] = (half_t)f.y;
            Bt[nv * 4 + 2][kr] = (half_t)f.z;
            Bt[nv * 4 + 3][kr] = (half_t)f.w;
        }
        __syncthreads();

        hv16 afr[2], bfr[4];
#pragma unroll
        for (int rt = 0; rt < 2; ++rt) {
            int m = wr * 32 + rt * 16 + lc;
            afr[rt] = cat8(*(const hv8*)&As[m][kb], *(const hv8*)&As[m][16 + kb]);
        }
#pragma unroll
        for (int ct = 0; ct < 4; ++ct) {
            int n = wc * 64 + ct * 16 + lc;
            bfr[ct] = cat8(*(const hv8*)&Bt[n][kb], *(const hv8*)&Bt[n][16 + kb]);
        }
#pragma unroll
        for (int rt = 0; rt < 2; ++rt)
#pragma unroll
            for (int ct = 0; ct < 4; ++ct)
                acc[rt][ct] = wmma16(afr[rt], bfr[ct], acc[rt][ct]);
        __syncthreads();
    }

    // epilogue: scatter into head-major Q/K/V f16
    const int rbase = (lane & 16) ? 8 : 0;
#pragma unroll
    for (int rt = 0; rt < 2; ++rt)
#pragma unroll
        for (int ct = 0; ct < 4; ++ct) {
            int gcol = n0 + wc * 64 + ct * 16 + lc;     // 0..1535
            int t = gcol >> 9;                           // 0=q 1=k 2=v
            int c = gcol & 511;
            int h = c >> 6, dd = c & 63;
            half_t* dst = (t == 0) ? q_ws : (t == 1) ? k_ws : v_ws;
            float bias = bqkv[gcol];
#pragma unroll
            for (int r = 0; r < 8; ++r) {
                int gm = m0 + wr * 32 + rt * 16 + rbase + r;
                int bb = gm / NQ;
                int nn = gm - bb * NQ;
                dst[(((size_t)bb * HEADS + h) * NQ + nn) * HDIM + dd] =
                    (half_t)(acc[rt][ct][r] + bias);
            }
        }
}

// ---------------------------------------------------------------------------
// Kernel 2: per-(b,h) window attention with Swin relative bias.
//   K tile staged by the Tensor Data Mover (TDM) with HW padding that
//   reproduces the 72-half LDS row stride; V staged manually (transposed).
// ---------------------------------------------------------------------------
__global__ __launch_bounds__(128) void attn_kernel(
    const half_t* __restrict__ q_ws, const half_t* __restrict__ k_ws,
    const half_t* __restrict__ v_ws, const float* __restrict__ bias_table,
    half_t* __restrict__ o_ws)
{
    __shared__ alignas(16) half_t Ks[NP][72];       // K row-major, padded stride
    __shared__ alignas(16) half_t Vt[HDIM][232];    // V transposed (dim-major)
    __shared__ alignas(16) half_t Ps[4][16][232];   // per-wave P scratch

    const int tid  = threadIdx.x;
    const int lane = tid & 31;
    const int wave = tid >> 5;
    const int bh = blockIdx.x;
    const int b  = bh >> 3;
    const int h  = bh & 7;

    const half_t* qg = q_ws + (size_t)bh * NQ * HDIM;
    const half_t* kg = k_ws + (size_t)bh * NQ * HDIM;
    const half_t* vg = v_ws + (size_t)bh * NQ * HDIM;

    // ---- TDM: DMA K[196][64] f16 -> Ks with row padding 128B + 16B ----
    if (wave == 0) {
        unsigned long long gaddr = (unsigned long long)(uintptr_t)kg;
        unsigned int lds_off = (unsigned int)(uintptr_t)(&Ks[0][0]); // low 32 = LDS byte offset
        uv4 g0;
        g0.x = 0x1u;                                   // count=1, user mode, no gather
        g0.y = lds_off;                                // lds_addr
        g0.z = (unsigned int)gaddr;                    // global_addr[31:0]
        g0.w = (unsigned int)((gaddr >> 32) & 0x01FFFFFFu) | (2u << 30); // addr[56:32] | type=2
        iv8 g1;
        g1[0] = (1 << 16)     // data_size = 2 bytes
              | (1 << 20)     // pad_enable
              | (4 << 22)     // pad_interval: 32 dwords (=128B row)
              | (3 << 25);    // pad_amount: 4 dwords (=16B -> stride 144B)
        g1[1] = (HDIM & 0xFFFF) << 16;                 // tensor_dim0[15:0]
        g1[2] = (NQ & 0xFFFF) << 16;                   // dim0 hi16=0 | tensor_dim1 lo16
        g1[3] = (HDIM << 16);                          // dim1 hi16=0 | tile_dim0
        g1[4] = NQ;                                    // tile_dim1 | tile_dim2=0
        g1[5] = HDIM;                                  // tensor_dim0_stride lo32
        g1[6] = 0;                                     // stride0 hi | stride1 lo
        g1[7] = 0;                                     // stride1 hi
        iv4 gz4 = {0, 0, 0, 0};
        iv8 gz8 = {0, 0, 0, 0, 0, 0, 0, 0};
        __builtin_amdgcn_tensor_load_to_lds(g0, g1, gz4, gz4, gz8, 0);
        __builtin_amdgcn_s_wait_tensorcnt(0);
    }
    // zero pad K rows 196..223 (disjoint from TDM region)
    for (int c = tid; c < (NP - NQ) * 8; c += 128) {
        int row = NQ + (c >> 3), cv = c & 7;
        hv8 z = {};
        *(hv8*)&Ks[row][cv * 8] = z;
    }
    // ---- stage V transposed (zero pad key cols 196..223) ----
    for (int c = tid; c < NQ * 8; c += 128) {
        int row = c >> 3, cv = c & 7;
        hv8 d = *(const hv8*)(vg + row * HDIM + cv * 8);
#pragma unroll
        for (int j = 0; j < 8; ++j) Vt[cv * 8 + j][row] = d[j];
    }
    for (int c = tid; c < HDIM * (NP - NQ); c += 128) {
        int dd = c / (NP - NQ);
        int n  = NQ + c % (NP - NQ);
        Vt[dd][n] = (half_t)0.f;
    }
    __syncthreads();

    const int kb    = (lane & 16) ? 8 : 0;
    const int koffb = (lane & 16);
    const int lc    = lane & 15;
    const int rbase = (lane & 16) ? 8 : 0;

    for (int qt = wave; qt < 13; qt += 4) {           // 13 query tiles cover 196 rows
        int m = qt * 16 + lc;
        bool mv = m < NQ;
        const half_t* qr = qg + (size_t)m * HDIM;
        hv8 z8 = {};
        hv8 a0lo = mv ? *(const hv8*)(qr + kb)      : z8;
        hv8 a0hi = mv ? *(const hv8*)(qr + 16 + kb) : z8;
        hv8 a1lo = mv ? *(const hv8*)(qr + 32 + kb) : z8;
        hv8 a1hi = mv ? *(const hv8*)(qr + 48 + kb) : z8;
        hv16 qa0 = cat8(a0lo, a0hi);
        hv16 qa1 = cat8(a1lo, a1hi);

        // S = Q K^T over 14 key tiles
        fv8 s[14];
#pragma unroll
        for (int kt = 0; kt < 14; ++kt) {
            int n = kt * 16 + lc;
            hv16 b0 = cat8(*(const hv8*)&Ks[n][kb],      *(const hv8*)&Ks[n][16 + kb]);
            hv16 b1 = cat8(*(const hv8*)&Ks[n][32 + kb], *(const hv8*)&Ks[n][48 + kb]);
            fv8 acc = {};
            acc = wmma16(qa0, b0, acc);
            acc = wmma16(qa1, b1, acc);
            s[kt] = acc;
        }

        // scale + relative-position bias + key mask (Swin index inline)
#pragma unroll
        for (int kt = 0; kt < 14; ++kt) {
            int n = kt * 16 + lc;
#pragma unroll
            for (int r = 0; r < 8; ++r) {
                int mm = qt * 16 + rbase + r;
                float v = s[kt][r];
                if (n < NQ && mm < NQ) {
                    int qy = mm / 14, qx = mm - qy * 14;
                    int ky = n / 14,  kx = n - ky * 14;
                    int idx = (qy - ky + 13) * 27 + (qx - kx + 13);
                    v = v * 0.125f + bias_table[idx * HEADS + h];
                } else {
                    v = -1e30f;
                }
                s[kt][r] = v;
            }
        }

        // row softmax: local over 14 tiles + cross-lane over each 16-lane half
        float inv[8];
#pragma unroll
        for (int r = 0; r < 8; ++r) {
            float mx = s[0][r];
#pragma unroll
            for (int kt = 1; kt < 14; ++kt) mx = fmaxf(mx, s[kt][r]);
#pragma unroll
            for (int msk = 1; msk < 16; msk <<= 1) mx = fmaxf(mx, __shfl_xor(mx, msk, 32));
            float sm = 0.f;
#pragma unroll
            for (int kt = 0; kt < 14; ++kt) {
                float p = __expf(s[kt][r] - mx);
                s[kt][r] = p;
                sm += p;
            }
#pragma unroll
            for (int msk = 1; msk < 16; msk <<= 1) sm += __shfl_xor(sm, msk, 32);
            inv[r] = 1.0f / sm;
        }

        // spill P to LDS (C layout -> row-major), reload as A fragments
#pragma unroll
        for (int kt = 0; kt < 14; ++kt)
#pragma unroll
            for (int r = 0; r < 8; ++r)
                Ps[wave][rbase + r][kt * 16 + lc] = (half_t)(s[kt][r] * inv[r]);

        fv8 o[4];
#pragma unroll
        for (int ct = 0; ct < 4; ++ct) { fv8 z = {}; o[ct] = z; }
#pragma unroll
        for (int kc = 0; kc < 7; ++kc) {              // 224 keys / 32
            const half_t* pr = &Ps[wave][lc][kc * 32];
            hv16 pa = cat8(*(const hv8*)(pr + kb), *(const hv8*)(pr + 16 + kb));
#pragma unroll
            for (int ct = 0; ct < 4; ++ct) {
                int n = ct * 16 + lc;
                const half_t* vr = &Vt[n][kc * 32];
                hv16 vb = cat8(*(const hv8*)(vr + kb), *(const hv8*)(vr + 16 + kb));
                o[ct] = wmma16(pa, vb, o[ct]);
            }
        }

        // write O into [BN, 512] with channel = h*64 + d
        half_t* ob = o_ws + (size_t)b * NQ * DMODEL + h * HDIM;
#pragma unroll
        for (int ct = 0; ct < 4; ++ct)
#pragma unroll
            for (int r = 0; r < 8; ++r) {
                int mm = qt * 16 + rbase + r;
                if (mm < NQ)
                    ob[(size_t)mm * DMODEL + ct * 16 + lc] = (half_t)o[ct][r];
            }
    }
}

// ---------------------------------------------------------------------------
// Kernel 3: out = O @ w_proj + b_proj   (f16 A, f32 B->f16, f32 out)
// ---------------------------------------------------------------------------
__global__ __launch_bounds__(256) void proj_kernel(
    const half_t* __restrict__ o_ws, const float* __restrict__ w,
    const float* __restrict__ bp, float* __restrict__ out)
{
    __shared__ alignas(16) half_t As[128][40];
    __shared__ alignas(16) half_t Bt[128][40];

    const int tid  = threadIdx.x;
    const int lane = tid & 31;
    const int wave = tid >> 5;
    const int wr = wave >> 1;
    const int wc = wave & 1;
    const int m0 = blockIdx.x * 128;
    const int n0 = blockIdx.y * 128;

    fv8 zero8 = {};
    fv8 acc[2][4];
#pragma unroll
    for (int rt = 0; rt < 2; ++rt)
#pragma unroll
        for (int ct = 0; ct < 4; ++ct) acc[rt][ct] = zero8;

    const int kb    = (lane & 16) ? 8 : 0;
    const int koffb = (lane & 16);
    const int lc    = lane & 15;

    for (int kk = 0; kk < DMODEL; kk += 32) {
        if (kk + 32 < DMODEL) {
            int arow = (tid * 2) >> 2;
            __builtin_prefetch(o_ws + (size_t)(m0 + arow) * DMODEL + kk + 32, 0, 1);
            int brow = (tid * 4) >> 5;
            __builtin_prefetch(w + (size_t)(kk + 32 + brow) * DMODEL + n0, 0, 1);
        }
        // stage A (already f16): 4096 halves = 512 hv8 slots
#pragma unroll
        for (int i = 0; i < 2; ++i) {
            int slot = tid + i * 256;
            int row = slot >> 2, cv = slot & 3;
            *(hv8*)&As[row][cv * 8] =
                *(const hv8*)(o_ws + (size_t)(m0 + row) * DMODEL + kk + cv * 8);
        }
        // stage B transposed
#pragma unroll
        for (int i = 0; i < 4; ++i) {
            int slot = tid + i * 256;
            int kr = slot >> 5, nv = slot & 31;
            const float4 f = *(const float4*)(w + (size_t)(kk + kr) * DMODEL + n0 + nv * 4);
            Bt[nv * 4 + 0][kr] = (half_t)f.x;
            Bt[nv * 4 + 1][kr] = (half_t)f.y;
            Bt[nv * 4 + 2][kr] = (half_t)f.z;
            Bt[nv * 4 + 3][kr] = (half_t)f.w;
        }
        __syncthreads();

        hv16 afr[2], bfr[4];
#pragma unroll
        for (int rt = 0; rt < 2; ++rt) {
            int m = wr * 32 + rt * 16 + lc;
            afr[rt] = cat8(*(const hv8*)&As[m][kb], *(const hv8*)&As[m][16 + kb]);
        }
#pragma unroll
        for (int ct = 0; ct < 4; ++ct) {
            int n = wc * 64 + ct * 16 + lc;
            bfr[ct] = cat8(*(const hv8*)&Bt[n][kb], *(const hv8*)&Bt[n][16 + kb]);
        }
#pragma unroll
        for (int rt = 0; rt < 2; ++rt)
#pragma unroll
            for (int ct = 0; ct < 4; ++ct)
                acc[rt][ct] = wmma16(afr[rt], bfr[ct], acc[rt][ct]);
        __syncthreads();
    }

    const int rbase = (lane & 16) ? 8 : 0;
#pragma unroll
    for (int rt = 0; rt < 2; ++rt)
#pragma unroll
        for (int ct = 0; ct < 4; ++ct) {
            int gcol = n0 + wc * 64 + ct * 16 + lc;
            float bias = bp[gcol];
#pragma unroll
            for (int r = 0; r < 8; ++r) {
                int gm = m0 + wr * 32 + rt * 16 + rbase + r;
                out[(size_t)gm * DMODEL + gcol] = acc[rt][ct][r] + bias;
            }
        }
}

// ---------------------------------------------------------------------------
extern "C" void kernel_launch(void* const* d_in, const int* in_sizes, int n_in,
                              void* d_out, int out_size, void* d_ws, size_t ws_size,
                              hipStream_t stream) {
    const float* x      = (const float*)d_in[0];
    const float* w_qkv  = (const float*)d_in[1];
    const float* b_qkv  = (const float*)d_in[2];
    const float* w_proj = (const float*)d_in[3];
    const float* b_proj = (const float*)d_in[4];
    const float* btab   = (const float*)d_in[5];
    float* out = (float*)d_out;

    // workspace carve (f16): Q,K,V [B,H,196,64] + O [BN,512]
    const size_t SEG = (size_t)BATCH * HEADS * NQ * HDIM;   // 51,380,224 halves
    half_t* q_ws = (half_t*)d_ws;
    half_t* k_ws = q_ws + SEG;
    half_t* v_ws = k_ws + SEG;
    half_t* o_ws = v_ws + SEG;

    qkv_kernel<<<dim3(MROWS / 128, 1536 / 128), 256, 0, stream>>>(
        x, w_qkv, b_qkv, q_ws, k_ws, v_ws);
    attn_kernel<<<dim3(BATCH * HEADS), 128, 0, stream>>>(
        q_ws, k_ws, v_ws, btab, o_ws);
    proj_kernel<<<dim3(MROWS / 128, DMODEL / 128), 256, 0, stream>>>(
        o_ws, w_proj, b_proj, out);
}
